// FANSNeuralOutputUpdate_120259084385
// MI455X (gfx1250) — compile-verified, matrix operand order checked by reading, not run on hardware
//
#include <hip/hip_runtime.h>

typedef __attribute__((ext_vector_type(16))) _Float16 v16h;
typedef __attribute__((ext_vector_type(8)))  _Float16 v8h;
typedef __attribute__((ext_vector_type(8)))  float    v8f;
typedef __attribute__((ext_vector_type(4)))  float    v4f;

#define NG   64   // groups (N_OUT)
#define HID  32   // hidden width H
#define NIN  32   // padded input features (gather folded into zeros)
#define KSEL 8    // selected features per group

__device__ __forceinline__ float fast_tanh(float x) {
#if __has_builtin(__builtin_amdgcn_tanhf)
  return __builtin_amdgcn_tanhf(x);          // v_tanh_f32 (gfx1250 transcendental)
#else
  float e = __expf(2.0f * x);
  return 1.0f - 2.0f / (e + 1.0f);
#endif
}

__global__ __launch_bounds__(256)
void fans_fused_kernel(const float* __restrict__ xf, const float* __restrict__ xb,
                       const int*   __restrict__ sel, const float* __restrict__ W1,
                       const float* __restrict__ b1,  const float* __restrict__ W2,
                       const float* __restrict__ b2,  float* __restrict__ out, int Bn)
{
  // [g][h][k] dense f16 weights with gather scattered in: 128 KB
  __shared__ __align__(16) _Float16 sW1[NG * HID * NIN];
  __shared__ __align__(16) float    sB1[NG * HID];   // 8 KB
  __shared__ __align__(16) float    sW2[NG * HID];   // 8 KB
  __shared__ __align__(16) float    sB2[NG];         // 256 B

  const int tid = threadIdx.x;

  // ---- cooperative build: zero, then scatter W1 through sel_idx ----
  uint4* zp = (uint4*)sW1;
  #pragma unroll
  for (int i = 0; i < (NG * HID * NIN * 2 / 16) / 256; ++i)
    zp[tid + i * 256] = make_uint4(0u, 0u, 0u, 0u);
  __syncthreads();

  for (int i = tid; i < NG * KSEL * HID; i += 256) {
    int h   = i & (HID - 1);
    int k   = (i >> 5) & (KSEL - 1);
    int g   = i >> 8;
    int kin = sel[g * KSEL + k];                       // which of the 32 inputs
    sW1[(g * HID + h) * NIN + kin] = (_Float16)W1[(g * KSEL + k) * HID + h];
  }
  for (int i = tid; i < NG * HID; i += 256) { sB1[i] = b1[i]; sW2[i] = W2[i]; }
  if (tid < NG) sB2[tid] = b2[tid];
  __syncthreads();

  // ---- per-wave setup: 32 batch rows (two 16-row WMMA tiles) ----
  const int lane = tid & 31;
  const int wave = tid >> 5;
  const int ln   = lane & 15;
  const int base = (lane >> 4) * 8;   // K-chunk base AND h sub-offset (ISA 16-bit layout)

  const long rowBase = ((long)blockIdx.x * 8 + wave) * 32;
  long r0 = rowBase + ln;        if (r0 >= Bn) r0 = Bn - 1;   // clamped loads keep EXEC uniform
  long r1 = rowBase + 16 + ln;   if (r1 >= Bn) r1 = Bn - 1;

  // B operand = x-tile, 32(K=feature) x 16(N=batch): K 0..15 from x_f, 16..31 from x_b
  v16h bx0, bx1;
  {
    const v4f* pf0 = (const v4f*)(xf + r0 * 16 + base);
    const v4f* pb0 = (const v4f*)(xb + r0 * 16 + base);
    const v4f* pf1 = (const v4f*)(xf + r1 * 16 + base);
    const v4f* pb1 = (const v4f*)(xb + r1 * 16 + base);
    v4f f0a = pf0[0], f0b = pf0[1], g0a = pb0[0], g0b = pb0[1];
    v4f f1a = pf1[0], f1b = pf1[1], g1a = pb1[0], g1b = pb1[1];
    #pragma unroll
    for (int j = 0; j < 4; ++j) {
      bx0[j]      = (_Float16)f0a[j];  bx0[4 + j]  = (_Float16)f0b[j];
      bx0[8 + j]  = (_Float16)g0a[j];  bx0[12 + j] = (_Float16)g0b[j];
      bx1[j]      = (_Float16)f1a[j];  bx1[4 + j]  = (_Float16)f1b[j];
      bx1[8 + j]  = (_Float16)g1a[j];  bx1[12 + j] = (_Float16)g1b[j];
    }
  }

  const bool doStore = (lane < 16);
  const long row0 = rowBase + ln;
  const long row1 = rowBase + 16 + ln;

  #pragma unroll 1
  for (int g = 0; g < NG; ++g) {
    // A operands = W1full[g] (M=h, K=feature), two 16-row h tiles
    const _Float16* wr0 = sW1 + (g * HID + ln) * NIN;
    const _Float16* wr1 = sW1 + (g * HID + 16 + ln) * NIN;
    v8h a0lo = *(const v8h*)(wr0 + base);
    v8h a0hi = *(const v8h*)(wr0 + 16 + base);
    v8h a1lo = *(const v8h*)(wr1 + base);
    v8h a1hi = *(const v8h*)(wr1 + 16 + base);
    v16h A0, A1;
    #pragma unroll
    for (int j = 0; j < 8; ++j) {
      A0[j] = a0lo[j];  A0[8 + j] = a0hi[j];
      A1[j] = a1lo[j];  A1[8 + j] = a1hi[j];
    }

    // accumulators seeded with b1 (bias depends only on h = M-row of D)
    const v4f* pb1a = (const v4f*)(sB1 + g * HID + base);
    const v4f* pb1b = (const v4f*)(sB1 + g * HID + 16 + base);
    v4f bia0 = pb1a[0], bia1 = pb1a[1], bib0 = pb1b[0], bib1 = pb1b[1];
    v8f c0, c1, d0, d1;
    #pragma unroll
    for (int r = 0; r < 4; ++r) {
      c0[r] = bia0[r]; c0[4 + r] = bia1[r];
      c1[r] = bib0[r]; c1[4 + r] = bib1[r];
      d0[r] = bia0[r]; d0[4 + r] = bia1[r];
      d1[r] = bib0[r]; d1[4 + r] = bib1[r];
    }

    // D[h, batch] = W1full[g] x x_tile + bias
    c0 = __builtin_amdgcn_wmma_f32_16x16x32_f16(false, A0, false, bx0, (short)0, c0, false, false);
    c1 = __builtin_amdgcn_wmma_f32_16x16x32_f16(false, A1, false, bx0, (short)0, c1, false, false);
    d0 = __builtin_amdgcn_wmma_f32_16x16x32_f16(false, A0, false, bx1, (short)0, d0, false, false);
    d1 = __builtin_amdgcn_wmma_f32_16x16x32_f16(false, A1, false, bx1, (short)0, d1, false, false);

    // layer 2: y = sum_h tanh(.) * W2[g,h]  — h lives in-lane (VGPR index + half)
    const v4f* pw2a = (const v4f*)(sW2 + g * HID + base);
    const v4f* pw2b = (const v4f*)(sW2 + g * HID + 16 + base);
    v4f w2a0 = pw2a[0], w2a1 = pw2a[1], w2b0 = pw2b[0], w2b1 = pw2b[1];

    float s0 = 0.f, s1 = 0.f;
    #pragma unroll
    for (int r = 0; r < 4; ++r) {
      s0 = fmaf(fast_tanh(c0[r]),     w2a0[r], s0);
      s0 = fmaf(fast_tanh(c0[4 + r]), w2a1[r], s0);
      s0 = fmaf(fast_tanh(c1[r]),     w2b0[r], s0);
      s0 = fmaf(fast_tanh(c1[4 + r]), w2b1[r], s0);
      s1 = fmaf(fast_tanh(d0[r]),     w2a0[r], s1);
      s1 = fmaf(fast_tanh(d0[4 + r]), w2a1[r], s1);
      s1 = fmaf(fast_tanh(d1[r]),     w2b0[r], s1);
      s1 = fmaf(fast_tanh(d1[4 + r]), w2b1[r], s1);
    }
    // combine the two lane-halves (h 0..7,16..23 vs 8..15,24..31)
    s0 += __shfl_xor(s0, 16, 32);
    s1 += __shfl_xor(s1, 16, 32);

    if (doStore) {
      float bb = sB2[g];
      if (row0 < Bn) out[row0 * NG + g] = s0 + bb;
      if (row1 < Bn) out[row1 * NG + g] = s1 + bb;
    }
  }
}

extern "C" void kernel_launch(void* const* d_in, const int* in_sizes, int n_in,
                              void* d_out, int out_size, void* d_ws, size_t ws_size,
                              hipStream_t stream) {
  const float* xf = (const float*)d_in[0];
  const float* xb = (const float*)d_in[1];
  const int*   sel = (const int*)d_in[2];
  const float* W1 = (const float*)d_in[3];
  const float* b1 = (const float*)d_in[4];
  const float* W2 = (const float*)d_in[5];
  const float* b2 = (const float*)d_in[6];
  float* out = (float*)d_out;

  int Bn = in_sizes[0] / 16;                 // batch size from x_f (B,16)
  int rowsPerBlock = 8 * 32;                 // 8 waves x 32 rows
  int blocks = (Bn + rowsPerBlock - 1) / rowsPerBlock;
  fans_fused_kernel<<<blocks, 256, 0, stream>>>(xf, xb, sel, W1, b1, W2, b2, out, Bn);
}
